// UnifiedGenerationHead_83004537962632
// MI455X (gfx1250) — matrix operand order repaired and checked by reading, not compiled, not run on hardware
//
#include <hip/hip_runtime.h>

// ---------------- types ----------------
typedef __attribute__((ext_vector_type(16))) __bf16 v16bf;
typedef __attribute__((ext_vector_type(8)))  __bf16 v8bf;
typedef __attribute__((ext_vector_type(8)))  float  v8f;
typedef __attribute__((ext_vector_type(4)))  int    v4i;

// async global->LDS (CDNA5). Guarded: fall back to sync copies if builtins absent.
#if defined(__HIP_DEVICE_COMPILE__) && \
    __has_builtin(__builtin_amdgcn_global_load_async_to_lds_b128) && \
    __has_builtin(__builtin_amdgcn_s_wait_asynccnt)
#define HAVE_ASYNC 1
#else
#define HAVE_ASYNC 0
#endif

// ---------------- problem constants ----------------
constexpr int B_   = 4;
constexpr int S_   = 1024;
constexpr int SMm_ = 1024;       // encoder length
constexpr int A_   = 512;
constexpr int D_   = 1024;
constexpr int V_   = 32000;
constexpr int L_   = 3;
constexpr int H_   = 16;
constexpr int DH_  = 64;         // head dim
constexpr int DFF_ = 4096;
constexpr int SMEM_ = SMm_ + 1;  // 1025 (y_ctx + encoder)
constexpr int SKP_  = 1056;      // padded cross-score row stride (mult of 32)
constexpr int TKP_  = 1056;      // padded token stride for V^T
constexpr int BS_   = B_ * S_;

#define BM 128
#define BN 128
#define BK 32
#define LSTR 40   // padded LDS row stride in elements

// ---------------- helpers ----------------
__device__ __forceinline__ void async_copy16(__bf16* lds, const __bf16* g)
{
#if HAVE_ASYNC
    typedef __attribute__((address_space(1))) v4i g_v4i;
    typedef __attribute__((address_space(3))) v4i l_v4i;
    __builtin_amdgcn_global_load_async_to_lds_b128(
        (g_v4i*)g, (l_v4i*)lds, 0, 0);
#else
    *reinterpret_cast<v8bf*>(lds) = *reinterpret_cast<const v8bf*>(g);
#endif
}

__device__ __forceinline__ void wait_async_4()
{
#if HAVE_ASYNC
    __builtin_amdgcn_s_wait_asynccnt(4);
#endif
}
__device__ __forceinline__ void wait_async_0()
{
#if HAVE_ASYNC
    __builtin_amdgcn_s_wait_asynccnt(0);
#endif
}

// unguarded stage of a 128x32 tile (rows contiguous-in-k, row stride ld)
__device__ __forceinline__ void stage_fast(const __bf16* __restrict__ G, long ld,
                                           int row0, int k0, __bf16* lds, int tid)
{
#pragma unroll
    for (int i = 0; i < 2; ++i) {
        int c  = tid + i * 256;       // 512 chunks of 8 elements
        int r  = c >> 2;
        int kc = (c & 3) * 8;
        async_copy16(lds + r * LSTR + kc, G + (long)(row0 + r) * ld + (k0 + kc));
    }
}

// guarded stage (rows < RV valid, k < K valid, zero-fill elsewhere)
__device__ __forceinline__ void stage_guarded(const __bf16* __restrict__ G, long ld,
                                              int row0, int k0, int RV, int K,
                                              __bf16* lds, int tid)
{
#pragma unroll
    for (int i = 0; i < 2; ++i) {
        int c    = tid + i * 256;
        int r    = c >> 2;
        int kc   = (c & 3) * 8;
        int grow = row0 + r;
        __bf16* dst = lds + r * LSTR + kc;
        if (grow < RV && (k0 + kc + 8) <= K) {
            *reinterpret_cast<v8bf*>(dst) =
                *reinterpret_cast<const v8bf*>(G + (long)grow * ld + (k0 + kc));
        } else {
#pragma unroll
            for (int j = 0; j < 8; ++j) {
                int kk = k0 + kc + j;
                dst[j] = (grow < RV && kk < K) ? G[(long)grow * ld + kk] : (__bf16)0.0f;
            }
        }
    }
}

// 8x WMMA on one staged tile pair
__device__ __forceinline__ void compute_tile(const __bf16* As, const __bf16* Bs,
                                             int wm, int wn, int lane, v8f (&acc)[2][4])
{
    const int half = lane >> 4;
    const int lr   = lane & 15;

    v16bf afrag[2];
#pragma unroll
    for (int r = 0; r < 2; ++r) {
        const __bf16* base = As + (wm + r * 16 + lr) * LSTR + half * 8;
        v8bf lo = *reinterpret_cast<const v8bf*>(base);       // K c0..c0+7
        v8bf hi = *reinterpret_cast<const v8bf*>(base + 16);  // K 16+c0..
        afrag[r] = __builtin_shufflevector(lo, hi, 0,1,2,3,4,5,6,7,8,9,10,11,12,13,14,15);
    }
    v16bf bfrag[4];
#pragma unroll
    for (int t = 0; t < 4; ++t) {
        const __bf16* base = Bs + (wn + t * 16 + lr) * LSTR + half * 8;
        v8bf lo = *reinterpret_cast<const v8bf*>(base);
        v8bf hi = *reinterpret_cast<const v8bf*>(base + 16);
        bfrag[t] = __builtin_shufflevector(lo, hi, 0,1,2,3,4,5,6,7,8,9,10,11,12,13,14,15);
    }
#pragma unroll
    for (int r = 0; r < 2; ++r)
#pragma unroll
        for (int t = 0; t < 4; ++t)
            acc[r][t] = __builtin_amdgcn_wmma_f32_16x16x32_bf16(
                false, afrag[r], false, bfrag[t], (short)0, acc[r][t], false, false);
}

// ---------------- GEMM: C = alpha * A x Bt^T + bias + resid ----------------
// A : M x K, row stride ldA, contiguous in k.
// Bt: N x K, row stride ldB, contiguous in k   (i.e. operand B pre-transposed).
// C : row stride ldC; optional fp32 (Cf) and/or bf16 (Cb) outputs.
// Batched via gridDim.z with two-level (z/nh, z%nh) strides.
__global__ __launch_bounds__(256)
void gemm_bf16_wmma(const __bf16* __restrict__ A, const __bf16* __restrict__ Bt,
                    const float* __restrict__ bias, const float* __restrict__ resid,
                    float* __restrict__ Cf, __bf16* __restrict__ Cb,
                    int M, int N, int K,
                    long ldA, long ldB, long ldC, int nh,
                    long sAb, long sAh, long sBb, long sBh, long sCb, long sCh,
                    float alpha, int relu)
{
    __shared__ __align__(16) __bf16 As[2][BM * LSTR];
    __shared__ __align__(16) __bf16 Bs[2][BN * LSTR];

    const int tid  = threadIdx.x;
    const int lane = tid & 31;
    const int wave = tid >> 5;
    const int wm = (wave & 3) * 32;
    const int wn = (wave >> 2) * 64;

    const int z  = blockIdx.z;
    const int zb = z / nh, zh = z % nh;
    A  += (long)zb * sAb + (long)zh * sAh;
    Bt += (long)zb * sBb + (long)zh * sBh;
    const long offC = (long)zb * sCb + (long)zh * sCh;

    const int m0 = blockIdx.y * BM;
    const int n0 = blockIdx.x * BN;

    v8f zero = {0.f, 0.f, 0.f, 0.f, 0.f, 0.f, 0.f, 0.f};
    v8f acc[2][4];
#pragma unroll
    for (int r = 0; r < 2; ++r)
#pragma unroll
        for (int t = 0; t < 4; ++t) acc[r][t] = zero;

    const bool interior = (m0 + BM <= M) && (n0 + BN <= N) && ((K & (BK - 1)) == 0);

    if (interior) {
        // double-buffered async pipeline
        stage_fast(A,  ldA, m0, 0, As[0], tid);
        stage_fast(Bt, ldB, n0, 0, Bs[0], tid);
        const int ktiles = K >> 5;
        for (int kt = 0; kt < ktiles; ++kt) {
            const int cur = kt & 1;
            if (kt + 1 < ktiles) {
                stage_fast(A,  ldA, m0, (kt + 1) * BK, As[cur ^ 1], tid);
                stage_fast(Bt, ldB, n0, (kt + 1) * BK, Bs[cur ^ 1], tid);
                wait_async_4();    // retire tile kt, leave kt+1 in flight
            } else {
                wait_async_0();
            }
            __syncthreads();
            compute_tile(As[cur], Bs[cur], wm, wn, lane, acc);
            __syncthreads();
        }
    } else {
        const int ktiles = (K + BK - 1) / BK;
        for (int kt = 0; kt < ktiles; ++kt) {
            const int k0 = kt * BK;
            stage_guarded(A,  ldA, m0, k0, M, K, As[0], tid);
            stage_guarded(Bt, ldB, n0, k0, N, K, Bs[0], tid);
            __syncthreads();
            compute_tile(As[0], Bs[0], wm, wn, lane, acc);
            __syncthreads();
        }
    }

    // ---- epilogue: VGPR i -> M = i + 8*(lane>=16), N = lane&15 ----
    const int cn = lane & 15;
    const int cm = (lane >> 4) * 8;
#pragma unroll
    for (int r = 0; r < 2; ++r) {
#pragma unroll
        for (int t = 0; t < 4; ++t) {
#pragma unroll
            for (int i = 0; i < 8; ++i) {
                int row = m0 + wm + r * 16 + cm + i;
                int col = n0 + wn + t * 16 + cn;
                if (row < M && col < N) {
                    float v = acc[r][t][i] * alpha;
                    long off = offC + (long)row * ldC + col;
                    if (bias)  v += bias[col];
                    if (resid) v += resid[off];
                    if (relu)  v = fmaxf(v, 0.f);
                    if (Cf) Cf[off] = v;
                    if (Cb) Cb[off] = (__bf16)v;
                }
            }
        }
    }
}

// ---------------- in-place masked softmax over bf16 score rows ----------------
__global__ __launch_bounds__(256)
void softmax_rows(__bf16* __restrict__ S, int Sq, int Sk, int stride, int causal)
{
    const long r = blockIdx.x;
    __bf16* row = S + r * (long)stride;
    const int qpos  = (int)(r % Sq);
    const int valid = causal ? (qpos + 1) : Sk;
    const int tid = threadIdx.x;

    float vals[5];
    float m = -1e30f;
#pragma unroll
    for (int i = 0; i < 5; ++i) {
        int j = tid + i * 256;
        float v = (j < valid) ? (float)row[j] : -1e30f;
        vals[i] = v;
        m = fmaxf(m, v);
    }
    __shared__ float redm[8], reds[8];
#pragma unroll
    for (int o = 16; o > 0; o >>= 1) m = fmaxf(m, __shfl_xor(m, o, 32));
    if ((tid & 31) == 0) redm[tid >> 5] = m;
    __syncthreads();
    m = redm[0];
#pragma unroll
    for (int i = 1; i < 8; ++i) m = fmaxf(m, redm[i]);

    float s = 0.f;
#pragma unroll
    for (int i = 0; i < 5; ++i) {
        int j = tid + i * 256;
        float e = (j < valid) ? __expf(vals[i] - m) : 0.f;
        vals[i] = e;
        s += e;
    }
#pragma unroll
    for (int o = 16; o > 0; o >>= 1) s += __shfl_xor(s, o, 32);
    if ((tid & 31) == 0) reds[tid >> 5] = s;
    __syncthreads();
    s = 0.f;
#pragma unroll
    for (int i = 0; i < 8; ++i) s += reds[i];
    float inv = 1.0f / s;
#pragma unroll
    for (int i = 0; i < 5; ++i) {
        int j = tid + i * 256;
        if (j < Sk) row[j] = (__bf16)(vals[i] * inv);   // masked tail -> 0
    }
}

// ---------------- layernorm (row = D elements), fp32 in, fp32 + bf16 out ----------------
__global__ __launch_bounds__(256)
void layernorm_rows(const float* __restrict__ X, const float* __restrict__ g,
                    const float* __restrict__ b, float* __restrict__ Yf,
                    __bf16* __restrict__ Yb, int D)
{
    const long r = blockIdx.x;
    const float* x = X + r * (long)D;
    const int tid = threadIdx.x;

    float v[4];
    float s = 0.f;
#pragma unroll
    for (int i = 0; i < 4; ++i) { v[i] = x[tid + i * 256]; s += v[i]; }

    __shared__ float red[8];
#pragma unroll
    for (int o = 16; o > 0; o >>= 1) s += __shfl_xor(s, o, 32);
    if ((tid & 31) == 0) red[tid >> 5] = s;
    __syncthreads();
    s = 0.f;
#pragma unroll
    for (int i = 0; i < 8; ++i) s += red[i];
    const float mean = s / (float)D;
    __syncthreads();

    float q = 0.f;
#pragma unroll
    for (int i = 0; i < 4; ++i) { float d = v[i] - mean; q += d * d; }
#pragma unroll
    for (int o = 16; o > 0; o >>= 1) q += __shfl_xor(q, o, 32);
    if ((tid & 31) == 0) red[tid >> 5] = q;
    __syncthreads();
    q = 0.f;
#pragma unroll
    for (int i = 0; i < 8; ++i) q += red[i];
    const float inv = rsqrtf(q / (float)D + 1e-5f);

#pragma unroll
    for (int i = 0; i < 4; ++i) {
        int c = tid + i * 256;
        float y = (v[i] - mean) * inv * g[c] + b[c];
        Yf[r * (long)D + c] = y;
        Yb[r * (long)D + c] = (__bf16)y;
    }
}

// ---------------- embedding ----------------
__global__ __launch_bounds__(256)
void embed_kernel(const int* __restrict__ ids, const float* __restrict__ tok,
                  const float* __restrict__ pos, float* __restrict__ Xf,
                  __bf16* __restrict__ Xb, int Sq, int D)
{
    const long bs = blockIdx.x;
    const int  s  = (int)(bs % Sq);
    const long id = ids[bs];
    const int tid = threadIdx.x;
#pragma unroll
    for (int i = 0; i < 4; ++i) {
        int d = tid + i * 256;
        float v = tok[id * (long)D + d] + pos[(long)s * D + d];
        Xf[bs * (long)D + d] = v;
        Xb[bs * (long)D + d] = (__bf16)v;
    }
}

// ---------------- f32 -> bf16 convert ----------------
__global__ __launch_bounds__(256)
void f32_to_bf16(const float* __restrict__ in, __bf16* __restrict__ out, long n)
{
    for (long i = (long)blockIdx.x * 256 + threadIdx.x; i < n; i += (long)gridDim.x * 256)
        out[i] = (__bf16)in[i];
}

// ---------------- f32 KxN -> bf16 NxK transposed convert (tiled) ----------------
__global__ __launch_bounds__(256)
void conv_transpose(const float* __restrict__ W, __bf16* __restrict__ Wt, int K, int N)
{
    __shared__ float tile[32][33];
    const long mat = blockIdx.z;
    const float* Wm  = W  + mat * (long)K * N;
    __bf16*      Wtm = Wt + mat * (long)K * N;
    const int kb = blockIdx.y * 32, nb = blockIdx.x * 32;
    const int tx = threadIdx.x & 31, ty = threadIdx.x >> 5;
#pragma unroll
    for (int i = 0; i < 4; ++i) {
        int k = kb + ty + i * 8, n = nb + tx;
        tile[ty + i * 8][tx] = (k < K && n < N) ? Wm[(long)k * N + n] : 0.f;
    }
    __syncthreads();
#pragma unroll
    for (int i = 0; i < 4; ++i) {
        int n = nb + ty + i * 8, k = kb + tx;
        if (n < N && k < K) Wtm[(long)n * K + k] = (__bf16)tile[tx][ty + i * 8];
    }
}

// ---------------- bf16 [Bt][T][D] -> [Bt][D][TP] transpose (zero-padded t) ----------------
__global__ __launch_bounds__(256)
void transpose_bf16(const __bf16* __restrict__ X, __bf16* __restrict__ Xt,
                    int T, int D, int TP)
{
    __shared__ float tile[32][33];
    const int b  = blockIdx.z;
    const int tb = blockIdx.y * 32, db = blockIdx.x * 32;
    const int tx = threadIdx.x & 31, ty = threadIdx.x >> 5;
    const __bf16* Xb  = X  + (long)b * T * D;
    __bf16*       Xtb = Xt + (long)b * D * TP;
#pragma unroll
    for (int i = 0; i < 4; ++i) {
        int t = tb + ty + i * 8, d = db + tx;
        tile[ty + i * 8][tx] = (t < T) ? (float)Xb[(long)t * D + d] : 0.f;
    }
    __syncthreads();
#pragma unroll
    for (int i = 0; i < 4; ++i) {
        int d = db + ty + i * 8, t = tb + tx;
        if (t < TP) Xtb[(long)d * TP + t] = (__bf16)tile[tx][ty + i * 8];
    }
}

// ---------------- scatter encoder_output into memory rows [1..SM] ----------------
__global__ __launch_bounds__(256)
void scatter_encoder(const float* __restrict__ enc, float* __restrict__ memf,
                     int SM, int D)
{
    const long per = (long)SM * D;
    const long n   = (long)B_ * per;
    for (long i = (long)blockIdx.x * 256 + threadIdx.x; i < n; i += (long)gridDim.x * 256) {
        long b = i / per, rem = i % per;
        memf[b * (long)(SM + 1) * D + D + rem] = enc[i];
    }
}

// ---------------- host-side launch ----------------
static inline void launch_gemm(hipStream_t st,
    const __bf16* A, const __bf16* Bt, const float* bias, const float* resid,
    float* Cf, __bf16* Cb, int M, int N, int K,
    long ldA, long ldB, long ldC,
    int nb, int nh, long sAb, long sAh, long sBb, long sBh, long sCb, long sCh,
    float alpha, int relu)
{
    dim3 g((N + BN - 1) / BN, (M + BM - 1) / BM, nb * nh);
    gemm_bf16_wmma<<<g, 256, 0, st>>>(A, Bt, bias, resid, Cf, Cb, M, N, K,
                                      ldA, ldB, ldC, nh,
                                      sAb, sAh, sBb, sBh, sCb, sCh, alpha, relu);
}

extern "C" void kernel_launch(void* const* d_in, const int* in_sizes, int n_in,
                              void* d_out, int out_size, void* d_ws, size_t ws_size,
                              hipStream_t stream)
{
    (void)in_sizes; (void)n_in; (void)out_size; (void)ws_size;

    const float* y_state = (const float*)d_in[0];
    const int*   ids     = (const int*)  d_in[1];
    const float* enc     = (const float*)d_in[2];
    const float* tok     = (const float*)d_in[3];
    const float* pos     = (const float*)d_in[4];
    const float* ctx_W   = (const float*)d_in[5];
    const float* ctx_b   = (const float*)d_in[6];
    const float* out_W   = (const float*)d_in[7];
    const float* out_b   = (const float*)d_in[8];
    const float* sa_W[4] = {(const float*)d_in[9],  (const float*)d_in[11],
                            (const float*)d_in[13], (const float*)d_in[15]};
    const float* sa_b[4] = {(const float*)d_in[10], (const float*)d_in[12],
                            (const float*)d_in[14], (const float*)d_in[16]};
    const float* ca_W[4] = {(const float*)d_in[17], (const float*)d_in[19],
                            (const float*)d_in[21], (const float*)d_in[23]};
    const float* ca_b[4] = {(const float*)d_in[18], (const float*)d_in[20],
                            (const float*)d_in[22], (const float*)d_in[24]};
    const float* ff_W1 = (const float*)d_in[25];
    const float* ff_b1 = (const float*)d_in[26];
    const float* ff_W2 = (const float*)d_in[27];
    const float* ff_b2 = (const float*)d_in[28];
    const float* ln_g[3] = {(const float*)d_in[29], (const float*)d_in[31], (const float*)d_in[33]};
    const float* ln_b[3] = {(const float*)d_in[30], (const float*)d_in[32], (const float*)d_in[34]};

    // ---- workspace bump allocator ----
    char* wp = (char*)d_ws;
    auto alloc = [&](size_t bytes) -> char* {
        char* r = wp;
        wp += (bytes + 255) & ~(size_t)255;
        return r;
    };
    // transposed bf16 weights (N x K layout)
    __bf16* wT_ctx = (__bf16*)alloc((size_t)A_ * D_ * 2);
    __bf16* wT_sa[4]; for (int i = 0; i < 4; ++i) wT_sa[i] = (__bf16*)alloc((size_t)L_ * D_ * D_ * 2);
    __bf16* wT_ca[4]; for (int i = 0; i < 4; ++i) wT_ca[i] = (__bf16*)alloc((size_t)L_ * D_ * D_ * 2);
    __bf16* wT_f1  = (__bf16*)alloc((size_t)L_ * D_ * DFF_ * 2);
    __bf16* wT_f2  = (__bf16*)alloc((size_t)L_ * DFF_ * D_ * 2);
    __bf16* wT_out = (__bf16*)alloc((size_t)D_ * V_ * 2);
    __bf16* yb     = (__bf16*)alloc((size_t)B_ * A_ * 2);
    float*  memf   = (float*) alloc((size_t)B_ * SMEM_ * D_ * 4);
    __bf16* memb   = (__bf16*)alloc((size_t)B_ * SMEM_ * D_ * 2);
    float*  xf     = (float*) alloc((size_t)BS_ * D_ * 4);
    __bf16* xb     = (__bf16*)alloc((size_t)BS_ * D_ * 2);
    float*  t1     = (float*) alloc((size_t)BS_ * D_ * 4);
    __bf16* qb     = (__bf16*)alloc((size_t)BS_ * D_ * 2);
    __bf16* kb     = (__bf16*)alloc((size_t)B_ * SMEM_ * D_ * 2);
    __bf16* vb     = (__bf16*)alloc((size_t)B_ * SMEM_ * D_ * 2);
    __bf16* vt     = (__bf16*)alloc((size_t)B_ * D_ * TKP_ * 2);   // V^T per batch, padded
    __bf16* ob     = (__bf16*)alloc((size_t)BS_ * D_ * 2);
    __bf16* sc     = (__bf16*)alloc((size_t)B_ * H_ * S_ * SKP_ * 2);
    __bf16* h1b    = (__bf16*)alloc((size_t)BS_ * DFF_ * 2);

    auto conv = [&](const float* src, __bf16* dst, long n) {
        int blocks = (int)((n + 255) / 256);
        if (blocks > 8192) blocks = 8192;
        f32_to_bf16<<<blocks, 256, 0, stream>>>(src, dst, n);
    };
    auto convT = [&](const float* src, __bf16* dst, int K, int N, int mats) {
        dim3 g((N + 31) / 32, (K + 31) / 32, mats);
        conv_transpose<<<g, 256, 0, stream>>>(src, dst, K, N);
    };

    // ---- weight conversion (transposed to N x K) ----
    convT(ctx_W, wT_ctx, A_, D_, 1);
    for (int i = 0; i < 4; ++i) convT(sa_W[i], wT_sa[i], D_, D_, L_);
    for (int i = 0; i < 4; ++i) convT(ca_W[i], wT_ca[i], D_, D_, L_);
    convT(ff_W1, wT_f1, D_, DFF_, L_);
    convT(ff_W2, wT_f2, DFF_, D_, L_);
    convT(out_W, wT_out, D_, V_, 1);
    conv(y_state, yb, (long)B_ * A_);

    // ---- memory = [y_ctx ; encoder_output] ----
    launch_gemm(stream, yb, wT_ctx, ctx_b, nullptr, memf, nullptr,
                B_, D_, A_, A_, A_, (long)SMEM_ * D_,
                1, 1, 0, 0, 0, 0, 0, 0, 1.0f, 0);
    scatter_encoder<<<8192, 256, 0, stream>>>(enc, memf, SMm_, D_);
    conv(memf, memb, (long)B_ * SMEM_ * D_);

    // ---- embedding ----
    embed_kernel<<<BS_, 256, 0, stream>>>(ids, tok, pos, xf, xb, S_, D_);

    for (int l = 0; l < L_; ++l) {
        const size_t wo = (size_t)l * D_ * D_;
        const float *bq = sa_b[0] + (size_t)l * D_, *bk = sa_b[1] + (size_t)l * D_;
        const float *bv = sa_b[2] + (size_t)l * D_, *bo = sa_b[3] + (size_t)l * D_;

        // ===== self attention =====
        launch_gemm(stream, xb, wT_sa[0] + wo, bq, nullptr, nullptr, qb, BS_, D_, D_,
                    D_, D_, D_, 1, 1, 0, 0, 0, 0, 0, 0, 1.0f, 0);
        launch_gemm(stream, xb, wT_sa[1] + wo, bk, nullptr, nullptr, kb, BS_, D_, D_,
                    D_, D_, D_, 1, 1, 0, 0, 0, 0, 0, 0, 1.0f, 0);
        launch_gemm(stream, xb, wT_sa[2] + wo, bv, nullptr, nullptr, vb, BS_, D_, D_,
                    D_, D_, D_, 1, 1, 0, 0, 0, 0, 0, 0, 1.0f, 0);
        transpose_bf16<<<dim3(D_ / 32, (TKP_ + 31) / 32, B_), 256, 0, stream>>>(
            vb, vt, S_, D_, TKP_);
        // scores[b,h] = (Q K^T)/8 : A=Q (ldA=D), Bt=K (ldB=D, contiguous in dh)
        launch_gemm(stream, qb, kb, nullptr, nullptr, nullptr, sc, S_, S_, DH_,
                    D_, D_, S_,
                    B_, H_, (long)S_ * D_, DH_, (long)S_ * D_, DH_,
                    (long)H_ * S_ * S_, (long)S_ * S_, 0.125f, 0);
        softmax_rows<<<B_ * H_ * S_, 256, 0, stream>>>(sc, S_, S_, S_, 1);
        // O[b,h] = P V : Bt = V^T (ldB = TKP, contiguous in tokens)
        launch_gemm(stream, sc, vt, nullptr, nullptr, nullptr, ob, S_, DH_, S_,
                    S_, TKP_, D_,
                    B_, H_, (long)H_ * S_ * S_, (long)S_ * S_,
                    (long)D_ * TKP_, (long)DH_ * TKP_,
                    (long)S_ * D_, DH_, 1.0f, 0);
        launch_gemm(stream, ob, wT_sa[3] + wo, bo, xf, t1, nullptr, BS_, D_, D_,
                    D_, D_, D_, 1, 1, 0, 0, 0, 0, 0, 0, 1.0f, 0);
        layernorm_rows<<<BS_, 256, 0, stream>>>(t1, ln_g[0] + (size_t)l * D_,
                                                ln_b[0] + (size_t)l * D_, xf, xb, D_);

        // ===== cross attention =====
        const float *cbq = ca_b[0] + (size_t)l * D_, *cbk = ca_b[1] + (size_t)l * D_;
        const float *cbv = ca_b[2] + (size_t)l * D_, *cbo = ca_b[3] + (size_t)l * D_;

        launch_gemm(stream, xb, wT_ca[0] + wo, cbq, nullptr, nullptr, qb, BS_, D_, D_,
                    D_, D_, D_, 1, 1, 0, 0, 0, 0, 0, 0, 1.0f, 0);
        launch_gemm(stream, memb, wT_ca[1] + wo, cbk, nullptr, nullptr, kb, B_ * SMEM_, D_, D_,
                    D_, D_, D_, 1, 1, 0, 0, 0, 0, 0, 0, 1.0f, 0);
        launch_gemm(stream, memb, wT_ca[2] + wo, cbv, nullptr, nullptr, vb, B_ * SMEM_, D_, D_,
                    D_, D_, D_, 1, 1, 0, 0, 0, 0, 0, 0, 1.0f, 0);
        transpose_bf16<<<dim3(D_ / 32, (TKP_ + 31) / 32, B_), 256, 0, stream>>>(
            vb, vt, SMEM_, D_, TKP_);
        launch_gemm(stream, qb, kb, nullptr, nullptr, nullptr, sc, S_, SMEM_, DH_,
                    D_, D_, SKP_,
                    B_, H_, (long)S_ * D_, DH_, (long)SMEM_ * D_, DH_,
                    (long)H_ * S_ * SKP_, (long)S_ * SKP_, 0.125f, 0);
        softmax_rows<<<B_ * H_ * S_, 256, 0, stream>>>(sc, S_, SMEM_, SKP_, 0);
        launch_gemm(stream, sc, vt, nullptr, nullptr, nullptr, ob, S_, DH_, SMEM_,
                    SKP_, TKP_, D_,
                    B_, H_, (long)H_ * S_ * SKP_, (long)S_ * SKP_,
                    (long)D_ * TKP_, (long)DH_ * TKP_,
                    (long)S_ * D_, DH_, 1.0f, 0);
        launch_gemm(stream, ob, wT_ca[3] + wo, cbo, xf, t1, nullptr, BS_, D_, D_,
                    D_, D_, D_, 1, 1, 0, 0, 0, 0, 0, 0, 1.0f, 0);
        layernorm_rows<<<BS_, 256, 0, stream>>>(t1, ln_g[1] + (size_t)l * D_,
                                                ln_b[1] + (size_t)l * D_, xf, xb, D_);

        // ===== feed-forward =====
        launch_gemm(stream, xb, wT_f1 + (size_t)l * D_ * DFF_, ff_b1 + (size_t)l * DFF_,
                    nullptr, nullptr, h1b, BS_, DFF_, D_,
                    D_, D_, DFF_, 1, 1, 0, 0, 0, 0, 0, 0, 1.0f, /*relu*/1);
        launch_gemm(stream, h1b, wT_f2 + (size_t)l * DFF_ * D_, ff_b2 + (size_t)l * D_,
                    xf, t1, nullptr, BS_, D_, DFF_,
                    DFF_, DFF_, D_, 1, 1, 0, 0, 0, 0, 0, 0, 1.0f, 0);
        layernorm_rows<<<BS_, 256, 0, stream>>>(t1, ln_g[2] + (size_t)l * D_,
                                                ln_b[2] + (size_t)l * D_, xf, xb, D_);
    }

    // ---- final vocab projection -> d_out (B,S,V) fp32 ----
    launch_gemm(stream, xb, wT_out, out_b, nullptr, (float*)d_out, nullptr,
                BS_, V_, D_, D_, D_, V_, 1, 1, 0, 0, 0, 0, 0, 0, 1.0f, 0);
}
